// VoxelResBackBone8xV3_85383949844635
// MI455X (gfx1250) — matrix operand order repaired and checked by reading, not compiled, not run on hardware
//
#include <hip/hip_runtime.h>

// ---------------------------------------------------------------------------
// MI455X (gfx1250) dense sparse-conv backbone.
// Matrix-core-bound plan: all activations (<80MB) stay L2-resident (192MB),
// so we feed v_wmma_f32_16x16x32_bf16 via implicit GEMM. One wave32 owns a
// 16-voxel x (NT*16)-Cout tile: a single A fragment (gathered activations) is
// reused against NT B fragments (packed weights). All NT B fragments are
// loaded before the NT WMMAs so loads issue as one clause and waits decay
// (loadcnt 6/4/2/0) instead of stalling each WMMA at loadcnt==0.
// All loads/stores use explicit addrspace(1) -> global_load/global_store
// (LOADcnt only, no flat/DScnt serialization). A-gather is branchless:
// unconditional load from a clamped address + data select.
// ---------------------------------------------------------------------------

typedef __attribute__((ext_vector_type(16))) __bf16   v16bf;
typedef __attribute__((ext_vector_type(8)))  float    v8f;
typedef __attribute__((ext_vector_type(4)))  unsigned u32x4;   // plain vector, no ctor

struct Geom {
  int Do, Ho, Wo;       // output grid
  int Di, Hi, Wi;       // input grid
  int kD, kH, kW;       // kernel extents
  int sz, sy, sx;       // strides
  int pz, py, px;       // pads (lo)
  int trans;            // 1 => lhs_dilation=2 transposed conv
};

// ---- addrspace(1) access helpers: force global_load/global_store ----------
__device__ __forceinline__ u32x4 gload16(const unsigned short* p) {
  return *(const __attribute__((address_space(1))) u32x4*)(unsigned long long)p;
}
__device__ __forceinline__ float gloadf(const float* p) {
  return *(const __attribute__((address_space(1))) float*)(unsigned long long)p;
}
__device__ __forceinline__ unsigned char gloadu8(const unsigned char* p) {
  return *(const __attribute__((address_space(1))) unsigned char*)(unsigned long long)p;
}
__device__ __forceinline__ unsigned short gloadu16(const unsigned short* p) {
  return *(const __attribute__((address_space(1))) unsigned short*)(unsigned long long)p;
}
__device__ __forceinline__ void gstoreu8(unsigned char* p, unsigned char v) {
  *(__attribute__((address_space(1))) unsigned char*)(unsigned long long)p = v;
}
__device__ __forceinline__ void gstoreu16(unsigned short* p, unsigned short v) {
  *(__attribute__((address_space(1))) unsigned short*)(unsigned long long)p = v;
}
__device__ __forceinline__ void gstoref(float* p, float v) {
  *(__attribute__((address_space(1))) float*)(unsigned long long)p = v;
}

__device__ __forceinline__ float bf2f(unsigned short h) {
  unsigned u = ((unsigned)h) << 16; float f; __builtin_memcpy(&f, &u, 4); return f;
}
__device__ __forceinline__ unsigned short f2bf(float f) {
  unsigned u; __builtin_memcpy(&u, &f, 4);
  u += 0x7FFFu + ((u >> 16) & 1u);            // round-to-nearest-even
  return (unsigned short)(u >> 16);
}

// per-tap input linear index; izb/iyb/ixb are the precomputed per-voxel bases
// (non-trans: o*stride-pad ; trans: o-pad). Division-free.
template <bool TRANS>
__device__ __forceinline__ int tap_il(int izb, int iyb, int ixb,
                                      int kd, int kh, int kw,
                                      int Di, int Hi, int Wi, int HiWi) {
  if constexpr (TRANS) {
    const int tz = izb + kd, ty = iyb + kh, tx = ixb + kw;
    const int o = tz | ty | tx;
    if (o < 0 || (o & 1)) return -1;          // negative or dilation hole
    const int iz = tz >> 1, iy = ty >> 1, ix = tx >> 1;
    if ((unsigned)iz >= (unsigned)Di) return -1;
    if ((unsigned)iy >= (unsigned)Hi) return -1;
    if ((unsigned)ix >= (unsigned)Wi) return -1;
    return iz * HiWi + iy * Wi + ix;
  } else {
    const int iz = izb + kd, iy = iyb + kh, ix = ixb + kw;
    if ((unsigned)iz >= (unsigned)Di) return -1;
    if ((unsigned)iy >= (unsigned)Hi) return -1;
    if ((unsigned)ix >= (unsigned)Wi) return -1;
    return iz * HiWi + iy * Wi + ix;
  }
}

struct ConvP {
  const unsigned short* X;     // bf16 activations [NvoxIn, Cin]
  const unsigned short* Wt;    // bf16 packed weights [Cout, Kpad]
  const float* scale;
  const float* bias;
  const unsigned char* mask;   // output-grid mask
  const unsigned short* resid; // bf16 residual [NvoxOut, Cout] or null
  unsigned short* Y;           // bf16 out (null if f32 out)
  float* Yf;                   // f32 out (final layer)
  Geom g;
  int Cout, Kpad, NvoxOut;
};

union Frag { v16bf v; u32x4 q[2]; };

template <int LC, int KD, int KH, int KW, bool TRANS, int NT>
__global__ __launch_bounds__(256) void conv3d_wmma(ConvP p) {
  constexpr int Cin = 1 << LC;
  constexpr int KT  = KD * KH * KW;

  const int lane   = threadIdx.x & 31;
  const int wave   = threadIdx.x >> 5;
  const int nTN    = p.Cout >> 4;
  const int nTNg   = nTN / NT;                    // N-tile groups (exact)
  const int nTM    = (p.NvoxOut + 15) >> 4;
  const int tile   = blockIdx.x * 8 + wave;       // 8 waves / 256-thread block
  if (tile >= nTM * nTNg) return;                 // wave-uniform: EXEC full for WMMA
  const int mT     = tile / nTNg;
  const int nTg    = tile - mT * nTNg;
  const int hi     = (lane >> 4) & 1;
  const int lanelo = lane & 15;

  // A-matrix row for this lane (clamped; out-of-range rows are never stored)
  int m = mT * 16 + lanelo;
  if (m >= p.NvoxOut) m = p.NvoxOut - 1;
  int izb, iyb, ixb;
  {
    const int hw = p.g.Ho * p.g.Wo;
    const int oz = m / hw; const int r2 = m - oz * hw;
    const int oy = r2 / p.g.Wo, ox = r2 - oy * p.g.Wo;
    if constexpr (TRANS) { izb = oz - p.g.pz; iyb = oy - p.g.py; ixb = ox - p.g.px; }
    else { izb = oz * p.g.sz - p.g.pz; iyb = oy * p.g.sy - p.g.py; ixb = ox * p.g.sx - p.g.px; }
  }
  const int HiWi = p.g.Hi * p.g.Wi;
  const int Di = p.g.Di, Hi = p.g.Hi, Wi = p.g.Wi;

  const int kb = hi ? 8 : 0;                      // A lane K sub-offset (ISA 7.12.2)
  const unsigned short* Bb[NT];
#pragma unroll
  for (int j = 0; j < NT; ++j)
    Bb[j] = p.Wt + (size_t)((nTg * NT + j) * 16 + lanelo) * p.Kpad + (hi ? 8 : 0);

  const v8f vz = {0.f, 0.f, 0.f, 0.f, 0.f, 0.f, 0.f, 0.f};
  v8f acc[NT];
#pragma unroll
  for (int j = 0; j < NT; ++j) acc[j] = vz;
  const u32x4 z4 = {0u, 0u, 0u, 0u};
  int koff = 0;                                   // walks packed-K in halfs

  if constexpr (Cin >= 32) {
    for (int kd = 0; kd < KD; ++kd)
      for (int kh = 0; kh < KH; ++kh)
#pragma unroll
        for (int kw = 0; kw < KW; ++kw) {
          const int il = tap_il<TRANS>(izb, iyb, ixb, kd, kh, kw, Di, Hi, Wi, HiWi);
          const bool valid = il >= 0;
          // clamped base: unconditional (safe) loads, branchless zero-select
          const unsigned short* Xp = p.X + (((size_t)(valid ? il : 0)) << LC);
          __builtin_prefetch(Bb[0] + koff + 256, 0, 1);  // global_prefetch_b8
#pragma unroll
          for (int cc = 0; cc < Cin; cc += 32) {
            const u32x4 q0 = gload16(Xp + cc + kb);
            const u32x4 q1 = gload16(Xp + cc + 16 + kb);
            Frag a;
            a.q[0] = valid ? q0 : z4;
            a.q[1] = valid ? q1 : z4;
            Frag b[NT];                           // batch all B loads -> one clause
#pragma unroll
            for (int j = 0; j < NT; ++j) {
              b[j].q[0] = gload16(Bb[j] + koff);
              b[j].q[1] = gload16(Bb[j] + koff + 16);
            }
#pragma unroll
            for (int j = 0; j < NT; ++j)
              acc[j] = __builtin_amdgcn_wmma_f32_16x16x32_bf16(
                  false, a.v, false, b[j].v, (short)0, acc[j], false, false);
            koff += 32;
          }
        }
  } else {
    // Cin == 16: each 32-K chunk spans two taps; fully unrolled so the
    // constant-divisor tap decomposition folds at compile time.
    constexpr int NCH = (KT * Cin + 31) / 32;     // 14 chunks for 27 taps
#pragma unroll
    for (int pc = 0; pc < NCH; ++pc) {
      const int t0 = 2 * pc, t1 = 2 * pc + 1;
      Frag a;
      {
        const int kd = t0 / (KH * KW), r = t0 % (KH * KW);
        const int il = tap_il<TRANS>(izb, iyb, ixb, kd, r / KW, r % KW, Di, Hi, Wi, HiWi);
        const bool v0 = il >= 0;
        const u32x4 q = gload16(p.X + (((size_t)(v0 ? il : 0)) << LC) + kb);
        a.q[0] = v0 ? q : z4;
      }
      if (t1 < KT) {
        const int kd = t1 / (KH * KW), r = t1 % (KH * KW);
        const int il = tap_il<TRANS>(izb, iyb, ixb, kd, r / KW, r % KW, Di, Hi, Wi, HiWi);
        const bool v1 = il >= 0;
        const u32x4 q = gload16(p.X + (((size_t)(v1 ? il : 0)) << LC) + kb);
        a.q[1] = v1 ? q : z4;
      } else {
        a.q[1] = z4;
      }
      Frag b[NT];
#pragma unroll
      for (int j = 0; j < NT; ++j) {
        b[j].q[0] = gload16(Bb[j] + koff);
        b[j].q[1] = gload16(Bb[j] + koff + 16);
      }
#pragma unroll
      for (int j = 0; j < NT; ++j)
        acc[j] = __builtin_amdgcn_wmma_f32_16x16x32_bf16(
            false, a.v, false, b[j].v, (short)0, acc[j], false, false);
      koff += 32;
    }
  }

  // Epilogue: D layout = lane holds col n = lane&15; rows r + 8*hi (VGPR r)
  const int rowBase = mT * 16 + hi * 8;
#pragma unroll
  for (int j = 0; j < NT; ++j) {
    const int   n  = (nTg * NT + j) * 16 + lanelo;
    const float sc = gloadf(p.scale + n);
    const float bb = gloadf(p.bias + n);
#pragma unroll
    for (int r = 0; r < 8; ++r) {
      const int vox = rowBase + r;
      if (vox >= p.NvoxOut) break;
      const float mk = gloadu8(p.mask + vox) ? 1.f : 0.f;
      float y = acc[j][r] * sc + bb;
      if (p.resid)                                         // SBB 2nd conv
        y = y * mk + bf2f(gloadu16(p.resid + (size_t)vox * p.Cout + n));
      y = fmaxf(y, 0.f) * mk;
      if (p.Yf) gstoref(p.Yf + (size_t)vox * p.Cout + n, y);
      else      gstoreu16(p.Y + (size_t)vox * p.Cout + n, f2bf(y));
    }
  }
}

// ---- weight repack: DHWIO f32 -> [Cout][Kpad] bf16 (K = tap*Cin + c) ------
__global__ void pack_weights(const float* __restrict__ W, unsigned short* __restrict__ Wt,
                             int KT, int Cin, int Cout, int Kpad) {
  const long i = (long)blockIdx.x * 256 + threadIdx.x;
  if (i >= (long)Cout * Kpad) return;
  const int co = (int)(i / Kpad);
  const int kk = (int)(i - (long)co * Kpad);
  unsigned short v = 0;
  if (kk < KT * Cin) {
    const int t = kk / Cin, c = kk - t * Cin;
    v = f2bf(gloadf(W + ((long)t * Cin + c) * Cout + co));
  }
  gstoreu16(Wt + i, v);
}

// ---- scatter voxel features into dense f32 grid + mask --------------------
__global__ void scatter_in(const float* __restrict__ vf, const int* __restrict__ coords,
                           float* __restrict__ Xin, unsigned char* __restrict__ m1, int nvox) {
  const int i = blockIdx.x * 256 + threadIdx.x;
  if (i >= nvox) return;
  const int z = coords[i * 4 + 1], y = coords[i * 4 + 2], x = coords[i * 4 + 3];
  const long lin = ((long)z * 192 + y) * 192 + x;
  for (int c = 0; c < 5; ++c) gstoref(Xin + lin * 5 + c, gloadf(vf + i * 5 + c));
  gstoreu8(m1 + lin, 1);
}

// ---- first layer: Cin=5 direct conv (tiny: 2.5 GF) ------------------------
__global__ void conv_ic0(const float* __restrict__ Xin, const float* __restrict__ W,
                         const float* __restrict__ s, const float* __restrict__ b,
                         const unsigned char* __restrict__ m1, unsigned short* __restrict__ Y) {
  const long i = (long)blockIdx.x * 256 + threadIdx.x;        // N1*16
  const long total = (long)16 * 192 * 192 * 16;
  if (i >= total) return;
  const int vox = (int)(i >> 4), co = (int)(i & 15);
  const int oz = vox / (192 * 192);
  const int r  = vox - oz * 192 * 192;
  const int oy = r / 192, ox = r - oy * 192;
  float acc = 0.f;
  for (int kd = 0; kd < 3; ++kd) {
    const int iz = oz - 1 + kd; if ((unsigned)iz >= 16u) continue;
    for (int kh = 0; kh < 3; ++kh) {
      const int iy = oy - 1 + kh; if ((unsigned)iy >= 192u) continue;
      for (int kw = 0; kw < 3; ++kw) {
        const int ix = ox - 1 + kw; if ((unsigned)ix >= 192u) continue;
        const long base = (((long)iz * 192 + iy) * 192 + ix) * 5;
        const int tap = (kd * 3 + kh) * 3 + kw;
        for (int c = 0; c < 5; ++c)
          acc += gloadf(Xin + base + c) * gloadf(W + ((long)tap * 5 + c) * 16 + co);
      }
    }
  }
  const float mk = gloadu8(m1 + vox) ? 1.f : 0.f;
  gstoreu16(Y + i, f2bf(fmaxf(acc * gloadf(s + co) + gloadf(b + co), 0.f) * mk));
}

// ---- mask max-pool (downmask) ---------------------------------------------
__global__ void downmask_k(const unsigned char* __restrict__ mi, unsigned char* __restrict__ mo,
                           Geom g, int NvoxOut) {
  const int vox = blockIdx.x * 256 + threadIdx.x;
  if (vox >= NvoxOut) return;
  const int hw = g.Ho * g.Wo;
  const int oz = vox / hw; const int r = vox - oz * hw;
  const int oy = r / g.Wo, ox = r - oy * g.Wo;
  const int HiWi = g.Hi * g.Wi;
  unsigned char any = 0;
  for (int kd = 0; kd < g.kD && !any; ++kd)
    for (int kh = 0; kh < g.kH && !any; ++kh)
      for (int kw = 0; kw < g.kW && !any; ++kw) {
        const int il = tap_il<false>(oz * g.sz - g.pz, oy * g.sy - g.py, ox * g.sx - g.px,
                                     kd, kh, kw, g.Di, g.Hi, g.Wi, HiWi);
        if (il >= 0 && gloadu8(mi + il)) any = 1;
      }
  gstoreu8(mo + vox, any);
}

// ---- elementwise bf16 add (hr = p1 + p2) ----------------------------------
__global__ void add_bf16(const unsigned short* __restrict__ a, const unsigned short* __restrict__ b,
                         unsigned short* __restrict__ o, long n) {
  const long i = (long)blockIdx.x * 256 + threadIdx.x;
  if (i >= n) return;
  gstoreu16(o + i, f2bf(bf2f(gloadu16(a + i)) + bf2f(gloadu16(b + i))));
}

// ===========================================================================
// Host side
// ===========================================================================
static Geom mkg(int Do, int Ho, int Wo, int Di, int Hi, int Wi,
                int kD, int kH, int kW, int sz, int sy, int sx,
                int pz, int py, int px, int trans) {
  Geom g; g.Do = Do; g.Ho = Ho; g.Wo = Wo; g.Di = Di; g.Hi = Hi; g.Wi = Wi;
  g.kD = kD; g.kH = kH; g.kW = kW; g.sz = sz; g.sy = sy; g.sx = sx;
  g.pz = pz; g.py = py; g.px = px; g.trans = trans; return g;
}

extern "C" void kernel_launch(void* const* d_in, const int* in_sizes, int n_in,
                              void* d_out, int out_size, void* d_ws, size_t ws_size,
                              hipStream_t stream) {
  (void)in_sizes; (void)n_in; (void)out_size; (void)ws_size;
  const int N1 = 16 * 192 * 192, N2 = 8 * 96 * 96, N3 = 4 * 48 * 48, N4 = 2 * 24 * 24;
  const int NOUT = 1 * 48 * 48;

  // deterministic bump allocator over d_ws
  unsigned char* bump = (unsigned char*)d_ws;
  auto alloc = [&](size_t bytes) -> unsigned char* {
    unsigned char* p = bump; bump += (bytes + 255) & ~(size_t)255; return p;
  };
  auto F = [&](int i) -> const float* { return (const float*)d_in[i]; };

  float*          Xin = (float*)alloc((size_t)N1 * 5 * 4);
  unsigned char*  m1  = alloc(N1);
  unsigned char*  m2  = alloc(N2);
  unsigned char*  m3  = alloc(N3);
  unsigned char*  m4  = alloc(N4);
  unsigned char*  mO  = alloc(NOUT);
  unsigned short* A1  = (unsigned short*)alloc((size_t)N1 * 16 * 2);
  unsigned short* B1  = (unsigned short*)alloc((size_t)N1 * 16 * 2);
  unsigned short* A2  = (unsigned short*)alloc((size_t)N2 * 32 * 2);
  unsigned short* B2  = (unsigned short*)alloc((size_t)N2 * 32 * 2);
  unsigned short* A3  = (unsigned short*)alloc((size_t)N3 * 64 * 2);
  unsigned short* B3  = (unsigned short*)alloc((size_t)N3 * 64 * 2);
  unsigned short* C3  = (unsigned short*)alloc((size_t)N3 * 64 * 2);
  unsigned short* A4  = (unsigned short*)alloc((size_t)N4 * 128 * 2);
  unsigned short* B4  = (unsigned short*)alloc((size_t)N4 * 128 * 2);

  auto packW = [&](const float* W, int KT, int Cin, int Cout) -> const unsigned short* {
    const int Kpad = ((KT * Cin + 31) / 32) * 32;
    unsigned short* dst = (unsigned short*)alloc((size_t)Cout * Kpad * 2);
    const long total = (long)Cout * Kpad;
    pack_weights<<<(unsigned)((total + 255) / 256), 256, 0, stream>>>(W, dst, KT, Cin, Cout, Kpad);
    return dst;
  };
  // sorted-pytree leaf indices: 0 vf, 1 coords, 2 batch, then params (keys
  // sorted; conv: b,s,w ; sbb: b1,b2,s1,s2,w1,w2)
  const unsigned short* w_convout  = packW(F(5),   3,  64,  64);
  const unsigned short* w_down2    = packW(F(8),  27,  16,  32);
  const unsigned short* w_down3    = packW(F(11), 27,  32,  64);
  const unsigned short* w_icr1a1   = packW(F(19), 27,  16,  16);
  const unsigned short* w_icr1a2   = packW(F(20), 27,  16,  16);
  const unsigned short* w_icr1b1   = packW(F(25), 27,  16,  16);
  const unsigned short* w_icr1b2   = packW(F(26), 27,  16,  16);
  const unsigned short* w_p1c      = packW(F(29), 27,  64,  64);
  const unsigned short* w_p1s1a    = packW(F(34), 27,  64,  64);
  const unsigned short* w_p1s1b    = packW(F(35), 27,  64,  64);
  const unsigned short* w_p1s2a    = packW(F(40), 27,  64,  64);
  const unsigned short* w_p1s2b    = packW(F(41), 27,  64,  64);
  const unsigned short* w_p1s2t11  = packW(F(44), 27,  64,  64);
  const unsigned short* w_p1s3a    = packW(F(49), 27,  64,  64);
  const unsigned short* w_p1s3b    = packW(F(50), 27,  64,  64);
  const unsigned short* w_p1t1     = packW(F(53), 27,  64,  64);
  const unsigned short* w_p2c      = packW(F(56), 27, 128, 128);
  const unsigned short* w_p2s2a    = packW(F(61), 27, 128, 128);
  const unsigned short* w_p2s2b    = packW(F(62), 27, 128, 128);
  const unsigned short* w_p2s2t22  = packW(F(65), 27, 128, 128);
  const unsigned short* w_p2s3a    = packW(F(70), 27, 128, 128);
  const unsigned short* w_p2s3b    = packW(F(71), 27, 128, 128);
  const unsigned short* w_p2t1     = packW(F(74), 27, 128,  64);
  const unsigned short* w_r2a1     = packW(F(79), 27,  32,  32);
  const unsigned short* w_r2a2     = packW(F(80), 27,  32,  32);
  const unsigned short* w_r2b1     = packW(F(85), 27,  32,  32);
  const unsigned short* w_r2b2     = packW(F(86), 27,  32,  32);
  const unsigned short* w_r3a1     = packW(F(91), 27,  64,  64);
  const unsigned short* w_r3a2     = packW(F(92), 27,  64,  64);
  const unsigned short* w_r3b1     = packW(F(97), 27,  64,  64);
  const unsigned short* w_r3b2     = packW(F(98), 27,  64,  64);
  const unsigned short* w_t11      = packW(F(101),27,  64,  64);
  const unsigned short* w_t12      = packW(F(104),27,  64, 128);

  const Geom gS1  = mkg(16,192,192, 16,192,192, 3,3,3, 1,1,1, 1,1,1, 0);
  const Geom gD2  = mkg( 8, 96, 96, 16,192,192, 3,3,3, 2,2,2, 1,1,1, 0);
  const Geom gS2  = mkg( 8, 96, 96,  8, 96, 96, 3,3,3, 1,1,1, 1,1,1, 0);
  const Geom gD3  = mkg( 4, 48, 48,  8, 96, 96, 3,3,3, 2,2,2, 1,1,1, 0);
  const Geom gS3  = mkg( 4, 48, 48,  4, 48, 48, 3,3,3, 1,1,1, 1,1,1, 0);
  const Geom gD4  = mkg( 2, 24, 24,  4, 48, 48, 3,3,3, 2,2,2, 1,1,1, 0);
  const Geom gS4  = mkg( 2, 24, 24,  2, 24, 24, 3,3,3, 1,1,1, 1,1,1, 0);
  const Geom gINV = mkg( 4, 48, 48,  2, 24, 24, 3,3,3, 1,1,1, 1,1,1, 1);
  const Geom gOUT = mkg( 1, 48, 48,  4, 48, 48, 3,1,1, 2,1,1, 0,0,0, 0);

  auto conv = [&](const unsigned short* X, const unsigned short* Wt,
                  const float* s, const float* b, const unsigned char* mk,
                  const unsigned short* resid, unsigned short* Y, float* Yf,
                  Geom g, int Cin, int Cout, int KT, int NvoxOut) {
    ConvP p; p.X = X; p.Wt = Wt; p.scale = s; p.bias = b; p.mask = mk; p.resid = resid;
    p.Y = Y; p.Yf = Yf; p.g = g; p.Cout = Cout;
    p.Kpad = ((KT * Cin + 31) / 32) * 32; p.NvoxOut = NvoxOut;
    const int nt = (Cout >= 64) ? 4 : (Cout == 32 ? 2 : 1);
    const int tiles = ((NvoxOut + 15) / 16) * ((Cout / 16) / nt);
    const unsigned blocks = (unsigned)((tiles + 7) / 8);
    if (g.trans)                       conv3d_wmma<7,3,3,3,true ,4><<<blocks,256,0,stream>>>(p);
    else if (Cin == 16 && nt == 1)     conv3d_wmma<4,3,3,3,false,1><<<blocks,256,0,stream>>>(p);
    else if (Cin == 16)                conv3d_wmma<4,3,3,3,false,2><<<blocks,256,0,stream>>>(p);
    else if (Cin == 32 && nt == 2)     conv3d_wmma<5,3,3,3,false,2><<<blocks,256,0,stream>>>(p);
    else if (Cin == 32)                conv3d_wmma<5,3,3,3,false,4><<<blocks,256,0,stream>>>(p);
    else if (Cin == 64 && g.kH == 3)   conv3d_wmma<6,3,3,3,false,4><<<blocks,256,0,stream>>>(p);
    else if (Cin == 64)                conv3d_wmma<6,3,1,1,false,4><<<blocks,256,0,stream>>>(p);
    else                               conv3d_wmma<7,3,3,3,false,4><<<blocks,256,0,stream>>>(p);
  };
  auto sbb = [&](unsigned short* x, unsigned short* t,
                 const unsigned short* w1, const unsigned short* w2, int base,
                 const unsigned char* mk, Geom g, int C, int Nv) {
    conv(x, w1, F(base + 2), F(base + 0), mk, nullptr, t, nullptr, g, C, C, 27, Nv);
    conv(t, w2, F(base + 3), F(base + 1), mk, x, x, nullptr, g, C, C, 27, Nv);
  };

  // ---- forward pass ----
  (void)hipMemsetAsync(Xin, 0, (size_t)N1 * 5 * 4, stream);
  (void)hipMemsetAsync(m1, 0, N1, stream);
  scatter_in<<<(30000 + 255) / 256, 256, 0, stream>>>(F(0), (const int*)d_in[1], Xin, m1, 30000);

  conv_ic0<<<(unsigned)(((long)N1 * 16 + 255) / 256), 256, 0, stream>>>(Xin, F(14), F(13), F(12), m1, A1);
  sbb(A1, B1, w_icr1a1, w_icr1a2, 15, m1, gS1, 16, N1);
  sbb(A1, B1, w_icr1b1, w_icr1b2, 21, m1, gS1, 16, N1);

  downmask_k<<<(N2 + 255) / 256, 256, 0, stream>>>(m1, m2, gD2, N2);
  conv(A1, w_down2, F(7), F(6), m2, nullptr, A2, nullptr, gD2, 16, 32, 27, N2);
  sbb(A2, B2, w_r2a1, w_r2a2, 75, m2, gS2, 32, N2);
  sbb(A2, B2, w_r2b1, w_r2b2, 81, m2, gS2, 32, N2);

  downmask_k<<<(N3 + 255) / 256, 256, 0, stream>>>(m2, m3, gD3, N3);
  conv(A2, w_down3, F(10), F(9), m3, nullptr, A3, nullptr, gD3, 32, 64, 27, N3);
  sbb(A3, B3, w_r3a1, w_r3a2, 87, m3, gS3, 64, N3);
  sbb(A3, B3, w_r3b1, w_r3b2, 93, m3, gS3, 64, N3);
  sbb(A3, B3, w_p1s1a, w_p1s1b, 30, m3, gS3, 64, N3);            // p1 = A3

  downmask_k<<<(N4 + 255) / 256, 256, 0, stream>>>(m3, m4, gD4, N4);
  conv(A3, w_t12, F(103), F(102), m4, nullptr, A4, nullptr, gD4, 64, 128, 27, N4);  // p2 = A4
  conv(A3, w_t11, F(100), F(99), m3, nullptr, B3, nullptr, gS3, 64, 64, 27, N3);    // p1 = B3

  sbb(B3, C3, w_p1s2a, w_p1s2b, 36, m3, gS3, 64, N3);            // p1 = B3
  sbb(A4, B4, w_p2s2a, w_p2s2b, 57, m4, gS4, 128, N4);           // p2 = A4
  conv(B3, w_p1s2t11, F(43), F(42), m3, nullptr, C3, nullptr, gS3, 64, 64, 27, N3); // p1 = C3
  conv(A4, w_p2s2t22, F(64), F(63), m4, nullptr, B4, nullptr, gS4, 128, 128, 27, N4); // p2 = B4
  sbb(C3, A3, w_p1s3a, w_p1s3b, 45, m3, gS3, 64, N3);            // p1 = C3
  sbb(B4, A4, w_p2s3a, w_p2s3b, 66, m4, gS4, 128, N4);           // p2 = B4
  conv(C3, w_p1c, F(28), F(27), m3, nullptr, A3, nullptr, gS3, 64, 64, 27, N3);     // p1 = A3
  conv(B4, w_p2c, F(55), F(54), m4, nullptr, A4, nullptr, gS4, 128, 128, 27, N4);   // p2 = A4
  conv(A3, w_p1t1, F(52), F(51), m3, nullptr, B3, nullptr, gS3, 64, 64, 27, N3);    // p1 = B3
  conv(A4, w_p2t1, F(73), F(72), m3, nullptr, C3, nullptr, gINV, 128, 64, 27, N3);  // p2up = C3

  add_bf16<<<(unsigned)(((long)N3 * 64 + 255) / 256), 256, 0, stream>>>(B3, C3, A3, (long)N3 * 64);

  downmask_k<<<(NOUT + 255) / 256, 256, 0, stream>>>(m3, mO, gOUT, NOUT);
  conv(A3, w_convout, F(4), F(3), mO, nullptr, nullptr, (float*)d_out, gOUT, 64, 64, 3, NOUT);
}
